// UserBehaviorEncoder_55499567399362
// MI455X (gfx1250) — matrix operand order repaired and checked by reading, not compiled, not run on hardware
//
#include <hip/hip_runtime.h>
#include <stdint.h>

typedef __attribute__((ext_vector_type(16))) __bf16        v16bf;
typedef __attribute__((ext_vector_type(8)))  float         v8f;
typedef __attribute__((ext_vector_type(4)))  unsigned int  u32x4;
typedef __attribute__((ext_vector_type(2)))  int           i32x2;

#define HDIM   256
#define NNODES 2048
#define BATCH  8
#define NEGINF -9.0e15f

#define AS1 __attribute__((address_space(1)))
#define AS3 __attribute__((address_space(3)))

#if defined(__HIP_DEVICE_COMPILE__) && \
    __has_builtin(__builtin_amdgcn_global_load_async_to_lds_b64) && \
    __has_builtin(__builtin_amdgcn_s_wait_asynccnt)
#define HAVE_ASYNC_LDS 1
#else
#define HAVE_ASYNC_LDS 0
#endif

__device__ __forceinline__ float bf2f(unsigned short h) {
    return __uint_as_float(((unsigned)h) << 16);
}
__device__ __forceinline__ unsigned short f2bf(float f) {
    unsigned u = __float_as_uint(f);
    unsigned r = u + 0x7FFFu + ((u >> 16) & 1u);   // round-to-nearest-even
    return (unsigned short)(r >> 16);
}

#if HAVE_ASYNC_LDS
// Issue an 8-byte async global->LDS copy (ASYNCcnt-tracked).
__device__ __forceinline__ void async_copy_b64(const int* g, int* l) {
    __builtin_amdgcn_global_load_async_to_lds_b64(
        (AS1 i32x2*)(uintptr_t)g,
        (AS3 i32x2*)(uint32_t)(uintptr_t)l,  // low 32 bits of generic = LDS offset
        0, 0);
}
#endif

// ---------------------------------------------------------------------------
// f32 -> bf16 conversion
// ---------------------------------------------------------------------------
__global__ void cvt_f32_bf16(const float* __restrict__ x,
                             unsigned short* __restrict__ y, int n) {
    int i = blockIdx.x * blockDim.x + threadIdx.x;
    if (i < n) y[i] = f2bf(x[i]);
}

// ---------------------------------------------------------------------------
// Generic bf16 GEMM:  C[M,Nn] = act(A[M,K] * W[Nn,K]^T + bias)
// One wave per 16x16 C tile, v_wmma_f32_16x16x32_bf16 over K.
// transposed=1 stores per-batch transposed: out[b][col][node]  (Nn x NNODES)
// ---------------------------------------------------------------------------
__global__ __launch_bounds__(256)
void gemm_bf16_wmma(const unsigned short* __restrict__ A,
                    const unsigned short* __restrict__ W,
                    const float* __restrict__ bias,
                    unsigned short* __restrict__ out,
                    int M, int Nn, int K, int relu, int transposed) {
    int wave = blockIdx.x * (blockDim.x >> 5) + (threadIdx.x >> 5);
    int lane = threadIdx.x & 31;
    int hlf  = lane >> 4;          // which lane-half (K interleave)
    int lm   = lane & 15;          // row (A) / col (B,C) within tile
    int tilesN = Nn >> 4;
    int m0 = (wave / tilesN) << 4;
    int n0 = (wave % tilesN) << 4;

    // A fragment (16x32 bf16): lane<16 -> K {0..7, 16..23}; lane>=16 -> +8
    const unsigned short* arow = A + (size_t)(m0 + lm) * K + (hlf << 3);
    // B fragment (32x16 bf16): lane<16 -> K {0..7, 16..23}; lane>=16 -> +8; col=lm
    const unsigned short* brow = W + (size_t)(n0 + lm) * K + (hlf << 3);

    v8f c = {};
    for (int k0 = 0; k0 < K; k0 += 32) {
        union { u32x4 q[2]; v16bf v; } af, bfm;
        af.q[0]  = *(const u32x4*)(arow + k0);
        af.q[1]  = *(const u32x4*)(arow + k0 + 16);
        bfm.q[0] = *(const u32x4*)(brow + k0);
        bfm.q[1] = *(const u32x4*)(brow + k0 + 16);
        c = __builtin_amdgcn_wmma_f32_16x16x32_bf16(false, af.v, false, bfm.v,
                                                    (short)0, c, false, false);
    }
    float bv = bias ? bias[n0 + lm] : 0.0f;
    #pragma unroll
    for (int r = 0; r < 8; ++r) {
        float v = c[r] + bv;
        if (relu) v = v > 0.0f ? v : 0.0f;
        int row = m0 + r + (hlf << 3);
        int col = n0 + lm;
        if (transposed) {
            int bb = row >> 11, node = row & (NNODES - 1);
            out[((size_t)bb * Nn + col) * NNODES + node] = f2bf(v);
        } else {
            out[(size_t)row * Nn + col] = f2bf(v);
        }
    }
}

// ---------------------------------------------------------------------------
// src[b,n] = Wh[b,n,:].a_src ; dst[b,n] = Wh[b,n,:].a_dst  (reads WhT layout)
// ---------------------------------------------------------------------------
__global__ __launch_bounds__(256)
void srcdst_kernel(const unsigned short* __restrict__ WhT,
                   const float* __restrict__ a,
                   float* __restrict__ src, float* __restrict__ dst) {
    int idx = blockIdx.x * blockDim.x + threadIdx.x;     // 0..B*N-1
    int b = idx >> 11;
    int n = idx & (NNODES - 1);
    const unsigned short* base = WhT + (size_t)b * HDIM * NNODES + n;
    float s = 0.0f, d = 0.0f;
    #pragma unroll 4
    for (int h = 0; h < HDIM; ++h) {
        float w = bf2f(base[(size_t)h * NNODES]);        // coalesced across n
        s += w * a[h];
        d += w * a[HDIM + h];
    }
    src[idx] = s;
    dst[idx] = d;
}

// ---------------------------------------------------------------------------
// Per-row softmax stats (max, sum) over masked leaky-relu logits. Wave/row.
// ---------------------------------------------------------------------------
__global__ __launch_bounds__(256)
void rowstats_kernel(const int* __restrict__ adj, const float* __restrict__ src,
                     const float* __restrict__ dst, float* __restrict__ mrow,
                     float* __restrict__ srow) {
    int row  = (blockIdx.x * blockDim.x + threadIdx.x) >> 5;   // b*N + i
    int lane = threadIdx.x & 31;
    const int*   arow = adj + (size_t)row * NNODES;
    const float* dr   = dst + (row & ~(NNODES - 1));
    float si = src[row];

    float m = -3.0e38f;
    for (int j = lane; j < NNODES; j += 32) {
        float e = si + dr[j];
        e = e > 0.0f ? e : 0.2f * e;
        e = arow[j] > 0 ? e : NEGINF;
        m = fmaxf(m, e);
    }
    for (int o = 16; o > 0; o >>= 1) m = fmaxf(m, __shfl_xor(m, o, 32));

    float s = 0.0f;
    for (int j = lane; j < NNODES; j += 32) {
        float e = si + dr[j];
        e = e > 0.0f ? e : 0.2f * e;
        e = arow[j] > 0 ? e : NEGINF;
        s += __expf(e - m);
    }
    for (int o = 16; o > 0; o >>= 1) s += __shfl_xor(s, o, 32);

    if (lane == 0) { mrow[row] = m; srow[row] = s; }
}

// ---------------------------------------------------------------------------
// h'[b, i0:i0+16, :] = P(16xN) * Wh(NxH).  Block: 16 rows x all 256 cols.
// Adjacency tile double-buffered into LDS via async global->LDS copies
// (ASYNCcnt) so the next tile's fetch overlaps the current tile's
// softmax + WMMA work.  Each of 8 waves owns 32 cols (two 16x16 C tiles).
// ---------------------------------------------------------------------------
__global__ __launch_bounds__(256)
void aggregate_kernel(const int* __restrict__ adj, const float* __restrict__ src,
                      const float* __restrict__ dst,
                      const float* __restrict__ mrow, const float* __restrict__ srow,
                      const unsigned short* __restrict__ WhT,
                      unsigned short* __restrict__ hout) {
    __shared__ int adjt[2][16 * 32];

    int b    = blockIdx.x >> 7;
    int i0   = (blockIdx.x & 127) << 4;
    int wave = threadIdx.x >> 5;
    int lane = threadIdx.x & 31;
    int hlf  = lane >> 4;
    int lm   = lane & 15;

    int rowg = (b << 11) + i0 + lm;                // this lane's P row
    float si  = src[rowg];
    float mi  = mrow[rowg];
    float inv = 1.0f / srow[rowg];
    const float* dstb = dst + (b << 11);

    int n0 = wave << 5;
    const unsigned short* B0 = WhT + ((size_t)b * HDIM + n0 + lm)      * NNODES + (hlf << 3);
    const unsigned short* B1 = WhT + ((size_t)b * HDIM + n0 + 16 + lm) * NNODES + (hlf << 3);

    // cooperative adjacency stage: 256 threads x int2 = 16x32 ints per tile
    int ar = threadIdx.x >> 4;
    int ac = (threadIdx.x & 15) << 1;
    const int* gadj = adj + (size_t)((b << 11) + i0 + ar) * NNODES + ac;
    int koff = hlf << 3;

    int cur = 0;
#if HAVE_ASYNC_LDS
    async_copy_b64(gadj, adjt[0] + ar * 32 + ac);      // prologue: tile 0
#endif

    v8f c0 = {}, c1 = {};
    for (int jt = 0; jt < 64; ++jt) {
        int j0 = jt << 5;
#if HAVE_ASYNC_LDS
        if (jt < 63) {
            // issue next tile into the other buffer, then wait for current
            async_copy_b64(gadj + j0 + 32, adjt[cur ^ 1] + ar * 32 + ac);
            __builtin_amdgcn_s_wait_asynccnt(1);       // tile jt landed (in-order)
        } else {
            __builtin_amdgcn_s_wait_asynccnt(0);
        }
        __syncthreads();                               // whole tile visible
#else
        __syncthreads();
        *(int2*)(adjt[0] + ar * 32 + ac) = *(const int2*)(gadj + j0);
        if (jt < 63) __builtin_prefetch(gadj + j0 + 32, 0, 1);
        __syncthreads();
#endif
        const int* arow = adjt[cur] + lm * 32;

        // P fragment in A-matrix element order: k = {koff..koff+7, koff+16..+23}
        float p[16];
        #pragma unroll
        for (int u = 0; u < 8; ++u) {
            int k = koff + u;
            float e = si + dstb[j0 + k];
            e = e > 0.0f ? e : 0.2f * e;
            e = arow[k] > 0 ? e : NEGINF;
            p[u] = __expf(e - mi) * inv;
            int k2 = k + 16;
            float e2 = si + dstb[j0 + k2];
            e2 = e2 > 0.0f ? e2 : 0.2f * e2;
            e2 = arow[k2] > 0 ? e2 : NEGINF;
            p[8 + u] = __expf(e2 - mi) * inv;
        }
        union { unsigned int w[8]; v16bf v; } pa;
        #pragma unroll
        for (int u = 0; u < 8; ++u)
            pa.w[u] = (unsigned)f2bf(p[2 * u]) | ((unsigned)f2bf(p[2 * u + 1]) << 16);

        union { u32x4 q[2]; v16bf v; } b0f, b1f;
        b0f.q[0] = *(const u32x4*)(B0 + j0);
        b0f.q[1] = *(const u32x4*)(B0 + j0 + 16);
        b1f.q[0] = *(const u32x4*)(B1 + j0);
        b1f.q[1] = *(const u32x4*)(B1 + j0 + 16);

        c0 = __builtin_amdgcn_wmma_f32_16x16x32_bf16(false, pa.v, false, b0f.v,
                                                     (short)0, c0, false, false);
        c1 = __builtin_amdgcn_wmma_f32_16x16x32_bf16(false, pa.v, false, b1f.v,
                                                     (short)0, c1, false, false);

#if HAVE_ASYNC_LDS
        __syncthreads();    // all reads of adjt[cur] done before next-iter issue
        cur ^= 1;
#endif
    }

    #pragma unroll
    for (int r = 0; r < 8; ++r) {
        int irow = i0 + r + (hlf << 3);
        size_t base = ((size_t)(b << 11) + irow) * HDIM;
        hout[base + n0 + lm]      = f2bf(c0[r]);
        hout[base + n0 + 16 + lm] = f2bf(c1[r]);
    }
}

// ---------------------------------------------------------------------------
// g = mean_i h ; out = relu(g W1^T + b1) W2^T + b2.  One block per batch.
// ---------------------------------------------------------------------------
__global__ __launch_bounds__(256)
void pool_mlp_kernel(const unsigned short* __restrict__ hbf,
                     const float* __restrict__ W1, const float* __restrict__ b1,
                     const float* __restrict__ W2, const float* __restrict__ b2,
                     float* __restrict__ out) {
    __shared__ float g[HDIM];
    __shared__ float y1[HDIM / 2];
    int b = blockIdx.x, t = threadIdx.x;
    const unsigned short* hb = hbf + (size_t)b * NNODES * HDIM;
    float s = 0.0f;
    for (int i = 0; i < NNODES; ++i) s += bf2f(hb[(size_t)i * HDIM + t]);
    g[t] = s * (1.0f / NNODES);
    __syncthreads();
    if (t < HDIM / 2) {
        float acc = b1[t];
        for (int h = 0; h < HDIM; ++h) acc += g[h] * W1[t * HDIM + h];
        y1[t] = acc > 0.0f ? acc : 0.0f;
    }
    __syncthreads();
    if (t < HDIM / 4) {
        float acc = b2[t];
        for (int k = 0; k < HDIM / 2; ++k) acc += y1[k] * W2[t * (HDIM / 2) + k];
        out[b * (HDIM / 4) + t] = acc;
    }
}

// ---------------------------------------------------------------------------
extern "C" void kernel_launch(void* const* d_in, const int* in_sizes, int n_in,
                              void* d_out, int out_size, void* d_ws, size_t ws_size,
                              hipStream_t stream) {
    (void)in_sizes; (void)n_in; (void)out_size; (void)ws_size;
    const float* node = (const float*)d_in[0];
    const int*   adjm = (const int*)  d_in[1];
    const float* Wemb = (const float*)d_in[2];
    const float* bemb = (const float*)d_in[3];
    const float* gatW = (const float*)d_in[4];
    const float* gata = (const float*)d_in[5];
    const float* W1   = (const float*)d_in[6];
    const float* b1   = (const float*)d_in[7];
    const float* W2   = (const float*)d_in[8];
    const float* b2   = (const float*)d_in[9];
    float* out = (float*)d_out;

    char* wsp = (char*)d_ws;
    size_t off = 0;
    auto alloc = [&](size_t bytes) -> void* {
        void* p = wsp + off;
        off += (bytes + 255) & ~(size_t)255;
        return p;
    };
    unsigned short* Xbf   = (unsigned short*)alloc((size_t)BATCH * NNODES * 128 * 2);
    unsigned short* h0    = (unsigned short*)alloc((size_t)BATCH * NNODES * HDIM * 2);
    unsigned short* h1    = (unsigned short*)alloc((size_t)BATCH * NNODES * HDIM * 2);
    unsigned short* WhT   = (unsigned short*)alloc((size_t)BATCH * HDIM * NNODES * 2);
    unsigned short* Wembb = (unsigned short*)alloc((size_t)HDIM * 128 * 2);
    unsigned short* gatWb = (unsigned short*)alloc((size_t)3 * HDIM * HDIM * 2);
    float* srcv = (float*)alloc((size_t)BATCH * NNODES * 4);
    float* dstv = (float*)alloc((size_t)BATCH * NNODES * 4);
    float* mrow = (float*)alloc((size_t)BATCH * NNODES * 4);
    float* srow = (float*)alloc((size_t)BATCH * NNODES * 4);

    int nX = BATCH * NNODES * 128;
    cvt_f32_bf16<<<(nX + 255) / 256, 256, 0, stream>>>(node, Xbf, nX);
    cvt_f32_bf16<<<(HDIM * 128 + 255) / 256, 256, 0, stream>>>(Wemb, Wembb, HDIM * 128);
    cvt_f32_bf16<<<(3 * HDIM * HDIM + 255) / 256, 256, 0, stream>>>(gatW, gatWb, 3 * HDIM * HDIM);

    const int Mrows = BATCH * NNODES;                        // 16384
    const int gemmBlocks = (Mrows / 16) * (HDIM / 16) / 8;   // 2048 (8 waves/block)

    // embed: h0 = relu(X W_emb^T + b_emb), row-major bf16
    gemm_bf16_wmma<<<gemmBlocks, 256, 0, stream>>>(Xbf, Wembb, bemb, h0,
                                                   Mrows, HDIM, 128, 1, 0);
    unsigned short* hc = h0;
    unsigned short* hn = h1;
    for (int l = 0; l < 3; ++l) {
        // WhT[b][h][node] = (hc gat_W[l]^T) transposed
        gemm_bf16_wmma<<<gemmBlocks, 256, 0, stream>>>(hc, gatWb + (size_t)l * HDIM * HDIM,
                                                       nullptr, WhT, Mrows, HDIM, HDIM, 0, 1);
        srcdst_kernel<<<Mrows / 256, 256, 0, stream>>>(WhT, gata + l * 2 * HDIM, srcv, dstv);
        rowstats_kernel<<<Mrows * 32 / 256, 256, 0, stream>>>(adjm, srcv, dstv, mrow, srow);
        aggregate_kernel<<<BATCH * (NNODES / 16), 256, 0, stream>>>(adjm, srcv, dstv,
                                                                    mrow, srow, WhT, hn);
        unsigned short* t = hc; hc = hn; hn = t;
    }
    pool_mlp_kernel<<<BATCH, 256, 0, stream>>>(hc, W1, b1, W2, b2, out);
}
